// PredCells_1185410974109
// MI455X (gfx1250) — matrix operand order repaired and checked by reading, not compiled, not run on hardware
//
#include <hip/hip_runtime.h>
#include <math.h>

// ---------------- configuration ----------------
#define NWG 64
#define TPB 256

typedef __attribute__((ext_vector_type(16))) __bf16 v16bf;
typedef __attribute__((ext_vector_type(8)))  float  v8f;

// ---------------- workspace layout (bytes) ----------------
// Tiled bf16 weights: each 16x32 tile = 1024B laid out in exact WMMA A-fragment
// order (two 512B halves: VGPR0-3 frags for lanes 0..31, then VGPR4-7 frags).
#define SZ_WCATM_1 (24576ull*1024ull)               /* [Wih|Whh] 4096x3072 per mid layer */
#define OFF_WCATM  0ull
#define OFF_WCATT  (OFF_WCATM + 4ull*SZ_WCATM_1)    /* [Wih_top|Whh_top] 4096x2048 */
#define SZ_WCATT   (16384ull*1024ull)
#define SZ_SQ_1    (2048ull*1024ull)                /* 1024x1024 tiled */
#define OFF_VMID   (OFF_WCATT + SZ_WCATT)           /* 4 x V_mid */
#define OFF_WMID   (OFF_VMID + 4ull*SZ_SQ_1)        /* 4 x W_mid */
#define OFF_W0     (OFF_WMID + 4ull*SZ_SQ_1)        /* 1024x64 (K padded 56->64) */
#define OFF_V1     (OFF_W0 + 131072ull)             /* 64x1024 (rows padded 56->64) */
// fp32 / bf16 state
#define OFF_C      (OFF_V1 + 131072ull)             /* c: 5*1024 f32 */
#define OFF_RECONR (OFF_C + 20480ull)               /* reconR: 4*1024 f32 */
#define OFF_RECON1 (OFF_RECONR + 16384ull)          /* recon1: 64 f32 */
#define OFF_XCATM  (OFF_RECON1 + 256ull)            /* x_cat per mid layer: 4*3072 bf16 = [BU|TD|h] */
#define OFF_XCATT  (OFF_XCATM + 24576ull)           /* x_cat top: 2048 bf16 = [BU|h] */
#define OFF_ZPART  (OFF_XCATT + 4096ull)            /* z partials: 4096*8 f32 */
#define OFF_LOSS   (OFF_ZPART + 131072ull)          /* loss parts: 256*17 f32 */
#define OFF_BAR    (OFF_LOSS + 17408ull)            /* barrier cnt @+0, gen @+256 */
#define OFF_END    (OFF_BAR + 512ull)               /* ~128.5 MiB total */

// ---------------- small device helpers ----------------
__device__ __forceinline__ unsigned short f2bf(float f) {
    unsigned u = __float_as_uint(f);
    unsigned r = ((u >> 16) & 1u) + 0x7fffu;        // round-to-nearest-even
    return (unsigned short)((u + r) >> 16);
}
__device__ __forceinline__ float sigm(float x) { return 1.f / (1.f + expf(-x)); }

__device__ __forceinline__ v8f wmma_bf16(v16bf a, v16bf b, v8f c) {
    return __builtin_amdgcn_wmma_f32_16x16x32_bf16(false, a, false, b, (short)0, c, false, false);
}

// B fragment: x-vector replicated across all 16 columns.
// 16-bit B (32x16): lanes 0..15 hold rows K=0..7,16..23 (2 per VGPR), lanes 16..31 rows 8..15,24..31.
// Works for global (xcat) and LDS (stage-A TD0) source pointers.
__device__ __forceinline__ v16bf bfrag(const unsigned short* xbf, int kt, int lane) {
    const uint4* xq = (const uint4*)(xbf + kt * 32);
    int h = (lane >> 4) & 1;
    union { uint4 q[2]; v16bf v; } b;
    b.q[0] = xq[h];
    b.q[1] = xq[h + 2];
    return b.v;
}

// A fragment from pre-swizzled tiled weights: tile = 1024B = 64 uint4 slots.
__device__ __forceinline__ v16bf afrag(const uint4* wq, long tileIdx, int lane) {
    const uint4* p = wq + tileIdx * 64;
    union { uint4 q[2]; v16bf v; } a;
    a.q[0] = p[lane];
    a.q[1] = p[lane + 32];
    return a.v;
}

// Device-wide sense-reversing barrier (all NWG blocks are co-resident).
__device__ __forceinline__ void gbar(unsigned* bar) {
    __syncthreads();
    if (threadIdx.x == 0) {
        unsigned* cnt = bar;
        unsigned* gen = bar + 64;   // separate cacheline
        __threadfence();            // release prior stores to agent scope
        unsigned g   = __atomic_load_n(gen, __ATOMIC_RELAXED);
        unsigned old = atomicAdd(cnt, 1u);
        if (old == NWG - 1) {
            __atomic_store_n(cnt, 0u, __ATOMIC_RELAXED);
            __atomic_store_n(gen, g + 1u, __ATOMIC_RELEASE);
        } else {
            while (__atomic_load_n(gen, __ATOMIC_ACQUIRE) == g)
                __builtin_amdgcn_s_sleep(2);
        }
    }
    __syncthreads();
    __threadfence();                // acquire: invalidate L0 before reading peers' data
}

// ---------------- init: zero all mutable state ----------------
__global__ void pc_init(unsigned int* dst, long nWords) {
    long i = (long)blockIdx.x * blockDim.x + threadIdx.x;
    long stride = (long)gridDim.x * blockDim.x;
    for (; i < nWords; i += stride) dst[i] = 0u;
}

// ---------------- fp32 -> bf16 tiled-fragment conversion ----------------
// Logical matrix (Rpad x Kpad) = [srcA (Rsrc x KA) | srcB (Rsrc x KB)], zero padded.
// One thread per 8-element (16B) fragment slot.
__global__ void pc_conv(const float* __restrict__ srcA, int KA,
                        const float* __restrict__ srcB, int KB,
                        int Rsrc, int RT, int KT, unsigned short* __restrict__ dst) {
    long g = (long)blockIdx.x * blockDim.x + threadIdx.x;
    long total = (long)RT * KT * 64;
    if (g >= total) return;
    long tile = g >> 6;
    int rem  = (int)(g & 63);
    int half = rem >> 5;
    int lane = rem & 31;
    int rt = (int)(tile / KT), kt = (int)(tile % KT);
    int row = rt * 16 + (lane & 15);
    int kb  = kt * 32 + half * 16 + ((lane >> 4) & 1) * 8;
    union { unsigned short s[8]; uint4 q; } out;
#pragma unroll
    for (int e = 0; e < 8; e++) {
        int k = kb + e;
        float f = 0.f;
        if (row < Rsrc) {
            if (k < KA)            f = srcA[(long)row * KA + k];
            else if (k < KA + KB)  f = srcB[(long)row * KB + (k - KA)];
        }
        out.s[e] = f2bf(f);
    }
    ((uint4*)dst)[g] = out.q;
}

// ---------------- persistent scan kernel ----------------
__global__ __launch_bounds__(TPB) void pc_main(
    const float* __restrict__ xin,      // [256,56]
    const float* __restrict__ W0_b,     // [1024]
    const float* __restrict__ Wmid_b,   // [4,1024]
    const float* __restrict__ V1_b,     // [56]
    const float* __restrict__ Vmid_b,   // [4,1024]
    const float* __restrict__ bmid,     // [4,4096]
    const float* __restrict__ btop,     // [4096]
    unsigned char* __restrict__ ws)
{
    float*          cbuf   = (float*)(ws + OFF_C);
    float*          reconR = (float*)(ws + OFF_RECONR);
    float*          recon1 = (float*)(ws + OFF_RECON1);
    unsigned short* xcatm  = (unsigned short*)(ws + OFF_XCATM);
    unsigned short* xcatt  = (unsigned short*)(ws + OFF_XCATT);
    float*          zpart  = (float*)(ws + OFF_ZPART);
    float*          loss   = (float*)(ws + OFF_LOSS);
    unsigned*       bar    = (unsigned*)(ws + OFF_BAR);

    const int tid  = blockIdx.x * TPB + threadIdx.x;
    const int wid  = tid >> 5;
    const int lane = threadIdx.x & 31;
    __shared__ float red[TPB];
    __shared__ unsigned short td0bf[64];   // stage-A TD0 in bf16, K padded 56->64

    for (int t = 0; t < 256; t++) {
        const float* xt = xin + t * 56;

        // ---- Stage A: BU0 = W0 @ |x_t - recon1| + W0_b (1024x64 matvec) ----
        if (blockIdx.x < 8) {               // wids 0..63 live in blocks 0..7
            // build TD0 bf16 once per block in LDS (uniform, no exec churn)
            if (threadIdx.x < 64) {
                int k = threadIdx.x;
                float f = (k < 56) ? fabsf(xt[k] - recon1[k]) : 0.f;
                td0bf[k] = f2bf(f);
            }
            __syncthreads();
            int jt = wid;                   // 0..63
            v8f acc = {0.f, 0.f, 0.f, 0.f, 0.f, 0.f, 0.f, 0.f};
#pragma unroll
            for (int kt = 0; kt < 2; kt++) {
                v16bf b = bfrag(td0bf, kt, lane);          // ds_load_b128 path
                v16bf a = afrag((const uint4*)(ws + OFF_W0), (long)jt * 2 + kt, lane);
                acc = wmma_bf16(a, b, acc);
            }
            int sel = (lane == 0) ? 0 : ((lane == 16) ? 8 : -1);
            if (sel >= 0) {
#pragma unroll
                for (int r = 0; r < 8; r++) {
                    int row = jt * 16 + sel + r;
                    xcatm[row] = f2bf(acc[r] + W0_b[row]);   // BU segment of layer-0 x_cat
                }
            }
        }
        if (blockIdx.x == 0) {  // loss part: sum |x_t - recon1|
            float v = (threadIdx.x < 56) ? fabsf(xt[threadIdx.x] - recon1[threadIdx.x]) : 0.f;
            red[threadIdx.x] = v; __syncthreads();
            for (int s = 128; s > 0; s >>= 1) {
                if (threadIdx.x < s) red[threadIdx.x] += red[threadIdx.x + s];
                __syncthreads();
            }
            if (threadIdx.x == 0) loss[t * 17] = red[0];
            __syncthreads();
        }
        gbar(bar);

        // ---- middle layers l = 1..4 (m = 0..3) ----
        for (int m = 0; m < 4; m++) {
            // Stage B: z = Wcat[m] @ [BU;TD;h] (4096x3072), 512 subunits (64 j-tiles x 8 K-slices)
            {
                int jt = wid >> 3, ks = wid & 7;
                const unsigned short* xv = xcatm + m * 3072;
                const uint4* wq = (const uint4*)(ws + OFF_WCATM + (size_t)m * SZ_WCATM_1);
                v8f acc0 = {0.f,0.f,0.f,0.f,0.f,0.f,0.f,0.f};
                v8f acc1 = acc0, acc2 = acc0, acc3 = acc0;
                for (int kt = ks * 12; kt < ks * 12 + 12; kt++) {
                    v16bf b = bfrag(xv, kt, lane);
                    __builtin_prefetch((const void*)(wq + ((long)jt * 96 + kt + 1) * 64), 0, 1);
                    acc0 = wmma_bf16(afrag(wq, (long)(0 * 64 + jt) * 96 + kt, lane), b, acc0);
                    acc1 = wmma_bf16(afrag(wq, (long)(1 * 64 + jt) * 96 + kt, lane), b, acc1);
                    acc2 = wmma_bf16(afrag(wq, (long)(2 * 64 + jt) * 96 + kt, lane), b, acc2);
                    acc3 = wmma_bf16(afrag(wq, (long)(3 * 64 + jt) * 96 + kt, lane), b, acc3);
                }
                int sel = (lane == 0) ? 0 : ((lane == 16) ? 8 : -1);
                if (sel >= 0) {
#pragma unroll
                    for (int r = 0; r < 8; r++) {
                        int rr = jt * 16 + sel + r;
                        zpart[((0 * 1024 + rr) << 3) + ks] = acc0[r];
                        zpart[((1 * 1024 + rr) << 3) + ks] = acc1[r];
                        zpart[((2 * 1024 + rr) << 3) + ks] = acc2[r];
                        zpart[((3 * 1024 + rr) << 3) + ks] = acc3[r];
                    }
                }
            }
            gbar(bar);

            // Stage G: gates + TD error (threads 0..1023)
            {
                float tdv = 0.f;
                if (tid < 1024) {
                    int j = tid;
                    float z[4];
#pragma unroll
                    for (int g = 0; g < 4; g++) {
                        const float4* p = (const float4*)(zpart + (((g << 10) + j) << 3));
                        float4 pa = p[0], pb = p[1];
                        float s = bmid[m * 4096 + (g << 10) + j];
                        s += pa.x; s += pa.y; s += pa.z; s += pa.w;
                        s += pb.x; s += pb.y; s += pb.z; s += pb.w;
                        z[g] = s;
                    }
                    float cj = cbuf[(m << 10) + j];
                    float cn = sigm(z[1]) * cj + sigm(z[0]) * tanhf(z[2]);
                    float hn = sigm(z[3]) * tanhf(cn);
                    cbuf[(m << 10) + j] = cn;
                    xcatm[m * 3072 + 2048 + j] = f2bf(hn);               // h segment
                    float td = fabsf(hn - reconR[(m << 10) + j]);
                    xcatm[m * 3072 + 1024 + j] = f2bf(td);               // TD segment
                    tdv = td;
                }
                if (blockIdx.x < 4) {                                    // loss part: sum TD
                    red[threadIdx.x] = tdv; __syncthreads();
                    for (int s = 128; s > 0; s >>= 1) {
                        if (threadIdx.x < s) red[threadIdx.x] += red[threadIdx.x + s];
                        __syncthreads();
                    }
                    if (threadIdx.x == 0) loss[t * 17 + 1 + m * 4 + blockIdx.x] = red[0];
                    __syncthreads();
                }
            }
            gbar(bar);

            // Stage C: recon = V @ h_new  and  BU_next = W_mid[m] @ TD  (1024x1024 each)
            {
                bool active = (wid < 128);
                bool isBU = wid >= 64;
                int  u = wid & 63;
                if (active && !isBU && m == 0 && u >= 4) active = false; // V1 has only 4 row-tiles
                if (active) {
                    const unsigned short* xv;
                    const uint4* wq;
                    if (isBU)       { xv = xcatm + m * 3072 + 1024; wq = (const uint4*)(ws + OFF_WMID + (size_t)m * SZ_SQ_1); }
                    else if (m == 0){ xv = xcatm + 2048;            wq = (const uint4*)(ws + OFF_V1); }
                    else            { xv = xcatm + m * 3072 + 2048; wq = (const uint4*)(ws + OFF_VMID + (size_t)(m - 1) * SZ_SQ_1); }
                    v8f acc = {0.f,0.f,0.f,0.f,0.f,0.f,0.f,0.f};
                    for (int kt = 0; kt < 32; kt++)
                        acc = wmma_bf16(afrag(wq, (long)u * 32 + kt, lane), bfrag(xv, kt, lane), acc);
                    int sel = (lane == 0) ? 0 : ((lane == 16) ? 8 : -1);
                    if (sel >= 0) {
#pragma unroll
                        for (int r = 0; r < 8; r++) {
                            int row = u * 16 + sel + r;
                            if (isBU) {
                                unsigned short hb = f2bf(acc[r] + Wmid_b[(m << 10) + row]);
                                if (m < 3) xcatm[(m + 1) * 3072 + row] = hb;
                                else       xcatt[row] = hb;
                            } else if (m == 0) {
                                if (row < 56) recon1[row] = acc[r] + V1_b[row];
                            } else {
                                reconR[((m - 1) << 10) + row] = acc[r] + Vmid_b[((m - 1) << 10) + row];
                            }
                        }
                    }
                }
            }
            gbar(bar);
        }

        // ---- top layer: z = Wcat_top @ [BU; h_top] (4096x2048) ----
        {
            int jt = wid >> 3, ks = wid & 7;
            const uint4* wq = (const uint4*)(ws + OFF_WCATT);
            v8f acc0 = {0.f,0.f,0.f,0.f,0.f,0.f,0.f,0.f};
            v8f acc1 = acc0, acc2 = acc0, acc3 = acc0;
            for (int kt = ks * 8; kt < ks * 8 + 8; kt++) {
                v16bf b = bfrag(xcatt, kt, lane);
                acc0 = wmma_bf16(afrag(wq, (long)(0 * 64 + jt) * 64 + kt, lane), b, acc0);
                acc1 = wmma_bf16(afrag(wq, (long)(1 * 64 + jt) * 64 + kt, lane), b, acc1);
                acc2 = wmma_bf16(afrag(wq, (long)(2 * 64 + jt) * 64 + kt, lane), b, acc2);
                acc3 = wmma_bf16(afrag(wq, (long)(3 * 64 + jt) * 64 + kt, lane), b, acc3);
            }
            int sel = (lane == 0) ? 0 : ((lane == 16) ? 8 : -1);
            if (sel >= 0) {
#pragma unroll
                for (int r = 0; r < 8; r++) {
                    int rr = jt * 16 + sel + r;
                    zpart[((0 * 1024 + rr) << 3) + ks] = acc0[r];
                    zpart[((1 * 1024 + rr) << 3) + ks] = acc1[r];
                    zpart[((2 * 1024 + rr) << 3) + ks] = acc2[r];
                    zpart[((3 * 1024 + rr) << 3) + ks] = acc3[r];
                }
            }
        }
        gbar(bar);

        // top gates
        if (tid < 1024) {
            int j = tid;
            float z[4];
#pragma unroll
            for (int g = 0; g < 4; g++) {
                const float4* p = (const float4*)(zpart + (((g << 10) + j) << 3));
                float4 pa = p[0], pb = p[1];
                float s = btop[(g << 10) + j];
                s += pa.x; s += pa.y; s += pa.z; s += pa.w;
                s += pb.x; s += pb.y; s += pb.z; s += pb.w;
                z[g] = s;
            }
            float cj = cbuf[(4 << 10) + j];
            float cn = sigm(z[1]) * cj + sigm(z[0]) * tanhf(z[2]);
            float hn = sigm(z[3]) * tanhf(cn);
            cbuf[(4 << 10) + j] = cn;
            xcatt[1024 + j] = f2bf(hn);
        }
        gbar(bar);

        // top recon: reconR[3] = V_mid[3] @ h_top + b
        if (wid < 64) {
            int u = wid;
            const unsigned short* xv = xcatt + 1024;
            const uint4* wq = (const uint4*)(ws + OFF_VMID + 3ull * SZ_SQ_1);
            v8f acc = {0.f,0.f,0.f,0.f,0.f,0.f,0.f,0.f};
            for (int kt = 0; kt < 32; kt++)
                acc = wmma_bf16(afrag(wq, (long)u * 32 + kt, lane), bfrag(xv, kt, lane), acc);
            int sel = (lane == 0) ? 0 : ((lane == 16) ? 8 : -1);
            if (sel >= 0) {
#pragma unroll
                for (int r = 0; r < 8; r++) {
                    int row = u * 16 + sel + r;
                    reconR[(3 << 10) + row] = acc[r] + Vmid_b[(3 << 10) + row];
                }
            }
        }
        gbar(bar);
    }
}

// ---------------- final deterministic loss reduction ----------------
__global__ void pc_final(const float* __restrict__ loss, float* __restrict__ out) {
    if (blockIdx.x == 0 && threadIdx.x == 0) {
        float s = 0.f;
        for (int i = 0; i < 256 * 17; i++) s += loss[i];
        out[0] = s;
    }
}

// ---------------- host launcher ----------------
extern "C" void kernel_launch(void* const* d_in, const int* in_sizes, int n_in,
                              void* d_out, int out_size, void* d_ws, size_t ws_size,
                              hipStream_t stream) {
    (void)in_sizes; (void)n_in; (void)out_size; (void)ws_size;
    const float* xin     = (const float*)d_in[0];
    const float* W0_w    = (const float*)d_in[1];
    const float* W0_b    = (const float*)d_in[2];
    const float* Wmid_w  = (const float*)d_in[3];
    const float* Wmid_b  = (const float*)d_in[4];
    const float* V1_w    = (const float*)d_in[5];
    const float* V1_b    = (const float*)d_in[6];
    const float* Vmid_w  = (const float*)d_in[7];
    const float* Vmid_b  = (const float*)d_in[8];
    const float* Wih_mid = (const float*)d_in[9];
    const float* Whh_mid = (const float*)d_in[10];
    const float* bmid    = (const float*)d_in[11];
    const float* Wih_top = (const float*)d_in[12];
    const float* Whh_top = (const float*)d_in[13];
    const float* btop    = (const float*)d_in[14];
    unsigned char* ws = (unsigned char*)d_ws;

    // zero mutable state (c, reconR, recon1, x_cat, zpart, loss, barrier)
    {
        long words = (long)((OFF_END - OFF_C) / 4);
        pc_init<<<64, TPB, 0, stream>>>((unsigned int*)(ws + OFF_C), words);
    }
    // fp32 -> tiled bf16 weight conversion (one-time per call; weights then live in L2)
    auto conv = [&](const float* A, int KA, const float* B, int KB,
                    int Rsrc, int RT, int KT, size_t off) {
        long slots = (long)RT * KT * 64;
        int blocks = (int)((slots + TPB - 1) / TPB);
        pc_conv<<<blocks, TPB, 0, stream>>>(A, KA, B, KB, Rsrc, RT, KT,
                                            (unsigned short*)(ws + off));
    };
    conv(W0_w, 56, nullptr, 0, 1024, 64, 2, OFF_W0);
    conv(V1_w, 1024, nullptr, 0, 56, 4, 32, OFF_V1);
    for (int m = 0; m < 4; m++) {
        conv(Wih_mid + (size_t)m * 4096 * 2048, 2048,
             Whh_mid + (size_t)m * 4096 * 1024, 1024,
             4096, 256, 96, OFF_WCATM + (size_t)m * SZ_WCATM_1);
        conv(Vmid_w + (size_t)m * 1024 * 1024, 1024, nullptr, 0,
             1024, 64, 32, OFF_VMID + (size_t)m * SZ_SQ_1);
        conv(Wmid_w + (size_t)m * 1024 * 1024, 1024, nullptr, 0,
             1024, 64, 32, OFF_WMID + (size_t)m * SZ_SQ_1);
    }
    conv(Wih_top, 1024, Whh_top, 1024, 4096, 256, 64, OFF_WCATT);

    // persistent scan
    pc_main<<<NWG, TPB, 0, stream>>>(xin, W0_b, Wmid_b, V1_b, Vmid_b, bmid, btop, ws);
    // deterministic final sum
    pc_final<<<1, 32, 0, stream>>>((const float*)(ws + OFF_LOSS), (float*)d_out);
}
